// LanguageModelSRNN_64682207478505
// MI455X (gfx1250) — matrix operand order repaired and checked by reading, not compiled, not run on hardware
//
#include <hip/hip_runtime.h>

// Problem dims (LanguageModelSRNN): B=8, S=1024, V=8000, H=256
#define NB   8
#define NS   1024
#define NV   8000
#define NH   256
#define ROWS (NB * NS)   // 8192
#define VPAD 8192        // padded vocab rows for B-matrix loads

typedef __attribute__((ext_vector_type(16))) __bf16 v16bf;
typedef __attribute__((ext_vector_type(8)))  __bf16 v8bf;
typedef __attribute__((ext_vector_type(8)))  float  v8f;

union V16U { v16bf v; v8bf h[2]; };

// ---------------------------------------------------------------------------
// Kernel 1: embedding gather  xp[row,:] = W_xh[X[row],:] + b_h
// ---------------------------------------------------------------------------
__global__ void srnn_embed(const int* __restrict__ X,
                           const float* __restrict__ Wxh,
                           const float* __restrict__ bh,
                           float* __restrict__ xp) {
    int row = blockIdx.x;          // 0..8191 (b*S + s)
    int h   = threadIdx.x;         // 0..255
    int tok = X[row];
    xp[(size_t)row * NH + h] = Wxh[(size_t)tok * NH + h] + bh[h];
}

// ---------------------------------------------------------------------------
// Kernel 2: W_out (H x V, f32) -> Wt (VPAD x H, bf16), zero-padded rows
// ---------------------------------------------------------------------------
__global__ void srnn_twout(const float* __restrict__ Wout,
                           __bf16* __restrict__ Wt) {
    int v = blockIdx.x;            // 0..8191
    int h = threadIdx.x;           // 0..255
    float val = (v < NV) ? Wout[(size_t)h * NV + v] : 0.0f;
    Wt[(size_t)v * NH + h] = (__bf16)val;
}

// ---------------------------------------------------------------------------
// Kernel 3: serial recurrence, one workgroup per batch.
// W_hh^T lives in LDS (f32, padded stride 260 -> ~266KB, fits CDNA5's 320KB).
// Double-buffered h => one barrier per timestep. Emits hs directly as bf16.
// ---------------------------------------------------------------------------
#define WST 260   // padded row stride (floats); 260*4 % 16 == 0, kills bank conflicts

__global__ void srnn_recur(const float* __restrict__ xp,
                           const float* __restrict__ Whh,
                           __bf16* __restrict__ hsb) {
    extern __shared__ float smem[];
    float* WT = smem;                    // [256][260]  WT[j][k] = Whh[k][j]
    float* h0 = smem + NH * WST;         // [256]
    float* h1 = h0 + NH;                 // [256]

    const int b = blockIdx.x;            // batch
    const int j = threadIdx.x;           // output unit

    // Stage W_hh transposed into LDS (coalesced global reads).
    for (int k = 0; k < NH; ++k)
        WT[j * WST + k] = Whh[(size_t)k * NH + j];
    h0[j] = 0.0f;
    h1[j] = 0.0f;
    __syncthreads();

    float* hcur = h0;
    float* hnxt = h1;
    const float* xprow = xp  + (size_t)(b * NS) * NH;
    __bf16*      hrow  = hsb + (size_t)(b * NS) * NH;

    for (int t = 0; t < NS; ++t) {
        float acc = xprow[(size_t)t * NH + j];
        const float4* wv = (const float4*)(WT + j * WST);
        const float4* hv = (const float4*)hcur;
        #pragma unroll 8
        for (int k4 = 0; k4 < NH / 4; ++k4) {
            float4 w = wv[k4];
            float4 hh = hv[k4];
            acc = fmaf(w.x, hh.x, acc);
            acc = fmaf(w.y, hh.y, acc);
            acc = fmaf(w.z, hh.z, acc);
            acc = fmaf(w.w, hh.w, acc);
        }
        float hn = tanhf(acc);
        hnxt[j] = hn;
        hrow[(size_t)t * NH + j] = (__bf16)hn;
        __syncthreads();
        float* tmp = hcur; hcur = hnxt; hnxt = tmp;
    }
}

// ---------------------------------------------------------------------------
// Kernel 4: logits = hs(8192x256) @ W_out(256x8000) + b_out  via bf16 WMMA.
// Block: 256 thr (8 waves, 2x4 grid). Block tile: 64(M) x 256(N).
// Wave tile: 32(M) x 64(N) = 2 M-tiles x 4 N-tiles = 8 f32 accumulators.
// K = 256 = 8 x (k32 WMMA steps) -> 64 WMMAs per wave; 8 WMMAs per 12 loads.
// ---------------------------------------------------------------------------
__global__ void srnn_gemm(const __bf16* __restrict__ Abf,  // ROWS x NH, row-major
                          const __bf16* __restrict__ Wt,   // VPAD x NH (W_out^T)
                          const float*  __restrict__ bout,
                          float* __restrict__ out) {
    const int lane = threadIdx.x & 31;
    const int wave = threadIdx.x >> 5;      // 0..7
    const int wm   = wave >> 2;             // 0..1 : M position in block
    const int wn   = wave & 3;              // 0..3 : N position in block

    const int m0 = blockIdx.x * 64 + wm * 32;   // two M-tiles: m0, m0+16
    const int n0 = blockIdx.y * 256 + wn * 64;  // four N-tiles

    const int msel  = lane & 15;
    const int hi    = lane >> 4;            // lane half
    const int akoff = hi * 16;              // A: contiguous 16 K per lane-half
    const int bkoff = hi * 16;              // B: contiguous 16 K per lane-half

    const __bf16* arow0 = Abf + (size_t)(m0 + msel) * NH;
    const __bf16* arow1 = arow0 + (size_t)16 * NH;

    v8f c00 = {}, c01 = {}, c02 = {}, c03 = {};
    v8f c10 = {}, c11 = {}, c12 = {}, c13 = {};

    #pragma unroll
    for (int ks = 0; ks < 8; ++ks) {
        const int kb = ks * 32;

        V16U a0, a1;
        a0.h[0] = *(const v8bf*)(arow0 + kb + akoff);
        a0.h[1] = *(const v8bf*)(arow0 + kb + akoff + 8);
        a1.h[0] = *(const v8bf*)(arow1 + kb + akoff);
        a1.h[1] = *(const v8bf*)(arow1 + kb + akoff + 8);

        const __bf16* bbase = Wt + (size_t)(n0 + msel) * NH + kb + bkoff;
        V16U b0, b1, b2, b3;
        b0.h[0] = *(const v8bf*)(bbase);
        b0.h[1] = *(const v8bf*)(bbase + 8);
        b1.h[0] = *(const v8bf*)(bbase + 16 * NH);
        b1.h[1] = *(const v8bf*)(bbase + 16 * NH + 8);
        b2.h[0] = *(const v8bf*)(bbase + 32 * NH);
        b2.h[1] = *(const v8bf*)(bbase + 32 * NH + 8);
        b3.h[0] = *(const v8bf*)(bbase + 48 * NH);
        b3.h[1] = *(const v8bf*)(bbase + 48 * NH + 8);

        c00 = __builtin_amdgcn_wmma_f32_16x16x32_bf16(false, a0.v, false, b0.v, (short)0, c00, false, false);
        c01 = __builtin_amdgcn_wmma_f32_16x16x32_bf16(false, a0.v, false, b1.v, (short)0, c01, false, false);
        c02 = __builtin_amdgcn_wmma_f32_16x16x32_bf16(false, a0.v, false, b2.v, (short)0, c02, false, false);
        c03 = __builtin_amdgcn_wmma_f32_16x16x32_bf16(false, a0.v, false, b3.v, (short)0, c03, false, false);
        c10 = __builtin_amdgcn_wmma_f32_16x16x32_bf16(false, a1.v, false, b0.v, (short)0, c10, false, false);
        c11 = __builtin_amdgcn_wmma_f32_16x16x32_bf16(false, a1.v, false, b1.v, (short)0, c11, false, false);
        c12 = __builtin_amdgcn_wmma_f32_16x16x32_bf16(false, a1.v, false, b2.v, (short)0, c12, false, false);
        c13 = __builtin_amdgcn_wmma_f32_16x16x32_bf16(false, a1.v, false, b3.v, (short)0, c13, false, false);
    }

    // C/D layout: VGPR r -> M = r + 8*hi, N = msel (per ISA 16x16 f32 table)
    v8f cc[2][4] = { {c00, c01, c02, c03}, {c10, c11, c12, c13} };
    #pragma unroll
    for (int t = 0; t < 4; ++t) {
        int col = n0 + t * 16 + msel;
        if (col < NV) {
            float bo = bout[col];
            #pragma unroll
            for (int mi = 0; mi < 2; ++mi) {
                #pragma unroll
                for (int r = 0; r < 8; ++r) {
                    int m = m0 + mi * 16 + hi * 8 + r;
                    out[(size_t)m * NV + col] = cc[mi][t][r] + bo;
                }
            }
        }
    }
}

// ---------------------------------------------------------------------------
// Kernel 5: in-place softmax over V per row (online max/sum, 2 passes).
// ---------------------------------------------------------------------------
__global__ void srnn_softmax(float* __restrict__ out) {
    __shared__ float sm[256];
    __shared__ float ss[256];
    const int row = blockIdx.x;
    const int tid = threadIdx.x;
    float* p = out + (size_t)row * NV;

    float m = -3.402823e38f, s = 0.0f;
    for (int i = tid; i < NV; i += 256) {
        float x  = p[i];
        float nm = fmaxf(m, x);
        s = s * __expf(m - nm) + __expf(x - nm);
        m = nm;
    }
    sm[tid] = m; ss[tid] = s;
    __syncthreads();
    for (int off = 128; off > 0; off >>= 1) {
        if (tid < off) {
            float m2 = sm[tid + off], s2 = ss[tid + off];
            float nm = fmaxf(sm[tid], m2);
            ss[tid] = ss[tid] * __expf(sm[tid] - nm) + s2 * __expf(m2 - nm);
            sm[tid] = nm;
        }
        __syncthreads();
    }
    const float M   = sm[0];
    const float inv = 1.0f / ss[0];
    for (int i = tid; i < NV; i += 256)
        p[i] = __expf(p[i] - M) * inv;
}

// ---------------------------------------------------------------------------
extern "C" void kernel_launch(void* const* d_in, const int* in_sizes, int n_in,
                              void* d_out, int out_size, void* d_ws, size_t ws_size,
                              hipStream_t stream) {
    const int*   X    = (const int*)  d_in[0];
    const float* Wxh  = (const float*)d_in[1];
    const float* Whh  = (const float*)d_in[2];
    const float* bh   = (const float*)d_in[3];
    const float* Wout = (const float*)d_in[4];
    const float* bout = (const float*)d_in[5];
    float* out = (float*)d_out;

    char* ws = (char*)d_ws;
    float*  xp  = (float*) (ws);                      //  8 MB: (8192,256) f32
    __bf16* hsb = (__bf16*)(ws +  8u * 1024 * 1024);  //  4 MB: (8192,256) bf16
    __bf16* Wt  = (__bf16*)(ws + 12u * 1024 * 1024);  //  4 MB: (8192,256) bf16 (W_out^T, zero-padded)

    srnn_embed<<<dim3(ROWS), dim3(NH), 0, stream>>>(X, Wxh, bh, xp);
    srnn_twout<<<dim3(VPAD), dim3(NH), 0, stream>>>(Wout, Wt);

    size_t smem = (size_t)(NH * WST + 2 * NH) * sizeof(float);  // ~268 KB < 320 KB/WGP
    srnn_recur<<<dim3(NB), dim3(NH), smem, stream>>>(xp, Whh, hsb);

    srnn_gemm<<<dim3(ROWS / 64, VPAD / 256), dim3(256), 0, stream>>>(hsb, Wt, bout, out);
    srnn_softmax<<<dim3(ROWS), dim3(256), 0, stream>>>(out);
}
